// AnonTokyoEncoder_20186346291286
// MI455X (gfx1250) — matrix operand order repaired and checked by reading, not compiled, not run on hardware
//
#include <hip/hip_runtime.h>
#include <hip/hip_bf16.h>
#include <math.h>

// ---------------------------------------------------------------------------
// Model constants (from the reference)
// ---------------------------------------------------------------------------
#define DMODEL   256
#define NHEADS   8
#define DHEAD    32
#define SPK      32          // SPARSE_K
#define BB       8           // batch
#define AA_N     64          // agents
#define TT       21          // timesteps
#define MM_N     768         // map polylines
#define PP       20          // points per polyline

#define AG_PTS   (BB * AA_N * TT)     // 10752
#define MP_PTS   (BB * MM_N * PP)     // 122880
#define AG_TOK   (BB * AA_N)          // 512
#define MP_TOK   (BB * MM_N)          // 6144

typedef unsigned short u16;
typedef __attribute__((ext_vector_type(16))) __bf16  v16bf;
typedef __attribute__((ext_vector_type(16))) u16     v16us;
typedef __attribute__((ext_vector_type(8)))  u16     v8us;
typedef __attribute__((ext_vector_type(8)))  float   v8f;

__device__ __forceinline__ u16 f2bf(float f) {
  unsigned u = __builtin_bit_cast(unsigned, f);
  u += 0x7FFFu + ((u >> 16) & 1u);          // round-to-nearest-even
  return (u16)(u >> 16);
}
__device__ __forceinline__ float bf2f(u16 h) {
  return __builtin_bit_cast(float, (unsigned)h << 16);
}

// ---------------------------------------------------------------------------
// Weight pre-pack: W[K,N] f32 row-major -> bf16 WMMA B-fragments, grouped so
// one wave streams a 64-wide N-strip contiguously:
//   out[(((strip*KB + k32)*4 + t4)*32 + lane)*16 + j]
// element j of lane's v16bf:  K = k32*32 + 16*(lane>>4) + j
//                             N = (strip*4 + t4)*16 + (lane&15)
// K zero-padded to a multiple of 32.  N must be a multiple of 64.
// ---------------------------------------------------------------------------
__global__ void pack_w_kernel(const float* __restrict__ W, int K, int N,
                              u16* __restrict__ out, int total)
{
  int i = blockIdx.x * blockDim.x + threadIdx.x;
  if (i >= total) return;
  int KB    = (K + 31) >> 5;
  int j     = i & 15;
  int lane  = (i >> 4) & 31;
  int t4    = (i >> 9) & 3;
  int t2    = i >> 11;
  int k32   = t2 % KB;
  int strip = t2 / KB;
  int kk  = k32 * 32 + 16 * (lane >> 4) + j;
  int col = (strip * 4 + t4) * 16 + (lane & 15);
  float v = (kk < K) ? W[(size_t)kk * N + col] : 0.f;
  out[i] = f2bf(v);
}

// ---------------------------------------------------------------------------
// WMMA GEMM:  out = relu?( [A1 | A2_bcast] * W + bias )
// One wave per block computes a 16(M) x 64(N) strip: per 32-K step it loads
// the A fragment once (2x b128) and 4 contiguous B fragments (8x b128), then
// issues 4 independent WMMAs sharing the A operand.
// K1, K2 multiples of 32; A1-blocks precede A2-blocks (no per-iter select).
// ---------------------------------------------------------------------------
#define GEMM_KSTEP(APTR)                                                     \
  {                                                                          \
    v8us alo = *(const v8us*)(APTR);                                         \
    v8us ahi = *(const v8us*)((APTR) + 8);                                   \
    v16us au = __builtin_shufflevector(alo, ahi,                             \
        0,1,2,3,4,5,6,7,8,9,10,11,12,13,14,15);                              \
    v16bf af = __builtin_bit_cast(v16bf, au);                                \
    v8us b0l = *(const v8us*)(wp);        v8us b0h = *(const v8us*)(wp + 8); \
    v8us b1l = *(const v8us*)(wp + 512);  v8us b1h = *(const v8us*)(wp + 520);\
    v8us b2l = *(const v8us*)(wp + 1024); v8us b2h = *(const v8us*)(wp + 1032);\
    v8us b3l = *(const v8us*)(wp + 1536); v8us b3h = *(const v8us*)(wp + 1544);\
    v16us bu0 = __builtin_shufflevector(b0l, b0h,                            \
        0,1,2,3,4,5,6,7,8,9,10,11,12,13,14,15);                              \
    v16us bu1 = __builtin_shufflevector(b1l, b1h,                            \
        0,1,2,3,4,5,6,7,8,9,10,11,12,13,14,15);                              \
    v16us bu2 = __builtin_shufflevector(b2l, b2h,                            \
        0,1,2,3,4,5,6,7,8,9,10,11,12,13,14,15);                              \
    v16us bu3 = __builtin_shufflevector(b3l, b3h,                            \
        0,1,2,3,4,5,6,7,8,9,10,11,12,13,14,15);                              \
    acc0 = __builtin_amdgcn_wmma_f32_16x16x32_bf16(false, af, false,         \
        __builtin_bit_cast(v16bf, bu0), (short)0, acc0, false, false);       \
    acc1 = __builtin_amdgcn_wmma_f32_16x16x32_bf16(false, af, false,         \
        __builtin_bit_cast(v16bf, bu1), (short)0, acc1, false, false);       \
    acc2 = __builtin_amdgcn_wmma_f32_16x16x32_bf16(false, af, false,         \
        __builtin_bit_cast(v16bf, bu2), (short)0, acc2, false, false);       \
    acc3 = __builtin_amdgcn_wmma_f32_16x16x32_bf16(false, af, false,         \
        __builtin_bit_cast(v16bf, bu3), (short)0, acc3, false, false);       \
    wp += 2048;                                                              \
  }

__global__ void __launch_bounds__(32) gemm_kernel(
    const u16* __restrict__ A1, int K1,
    const u16* __restrict__ A2, int K2, int rowDiv,
    const u16* __restrict__ Wp, const float* __restrict__ bias,
    float* __restrict__ outF, u16* __restrict__ outH,
    int N, int relu)
{
  const int lane  = threadIdx.x & 31;
  const int tileM = blockIdx.x;
  const int strip = blockIdx.y;          // 64 output columns
  const int half  = lane >> 4;
  const int l15   = lane & 15;
  const int rowA  = tileM * 16 + l15;
  const int KB    = (K1 + K2) >> 5;
  const int KB1   = K1 >> 5;
  const int KB2   = K2 >> 5;

  const u16* wp = Wp + ((size_t)strip * KB * 4 + 0) * 512 + lane * 16;

  v8f acc0 = {}, acc1 = {}, acc2 = {}, acc3 = {};

  const u16* ap = A1 + (size_t)rowA * K1 + 16 * half;
  for (int kb = 0; kb < KB1; ++kb) {
    GEMM_KSTEP(ap);
    ap += 32;
  }
  if (KB2) {
    const u16* ap2 = A2 + (size_t)(rowA / rowDiv) * K2 + 16 * half;
    for (int kb = 0; kb < KB2; ++kb) {
      GEMM_KSTEP(ap2);
      ap2 += 32;
    }
  }

#pragma unroll
  for (int t4 = 0; t4 < 4; ++t4) {
    v8f acc = (t4 == 0) ? acc0 : (t4 == 1) ? acc1 : (t4 == 2) ? acc2 : acc3;
    int colB = strip * 64 + t4 * 16 + l15;
    float bv = bias ? bias[colB] : 0.f;
#pragma unroll
    for (int i = 0; i < 8; ++i) {
      float v = acc[i] + bv;
      if (relu) v = fmaxf(v, 0.f);
      int r = tileM * 16 + i + 8 * half;
      if (outF) outF[(size_t)r * N + colB] = v;
      if (outH) outH[(size_t)r * N + colB] = f2bf(v);
    }
  }
}

// ---------------------------------------------------------------------------
// Masked max pooling over P points per group (bf16 in, bf16 out):
//   out[g,c] = any(mask) ? max_p(mask ? f : -1e9) : 0
// ---------------------------------------------------------------------------
__global__ void masked_max_kernel(const u16* __restrict__ f,
                                  const float* __restrict__ mask,
                                  u16* __restrict__ out, int G, int P, int C)
{
  int i = blockIdx.x * blockDim.x + threadIdx.x;
  if (i >= G * C) return;
  int g = i / C, c = i % C;
  float m = -1e9f; bool any = false;
  for (int p = 0; p < P; ++p) {
    bool valid = mask[(size_t)g * P + p] > 0.5f;
    float v = valid ? bf2f(f[((size_t)g * P + p) * C + c]) : -1e9f;
    m = fmaxf(m, v);
    any = any || valid;
  }
  out[(size_t)g * C + c] = f2bf(any ? m : 0.f);
}

// agent_input = concat(obj_trajs[...,10], mask[...,None]) -> bf16, K padded 32
__global__ void agent_input_kernel(const float* __restrict__ traj,
                                   const float* __restrict__ mask,
                                   u16* __restrict__ out, int Npts)
{
  int i = blockIdx.x * blockDim.x + threadIdx.x;
  if (i >= Npts * 32) return;
  int p = i >> 5, c = i & 31;
  float v = (c < 10) ? traj[(size_t)p * 10 + c] : ((c == 10) ? mask[p] : 0.f);
  out[i] = f2bf(v);
}

// map_polylines (K=7) -> bf16, K padded to 32
__global__ void map_input_kernel(const float* __restrict__ poly,
                                 u16* __restrict__ out, int Npts)
{
  int i = blockIdx.x * blockDim.x + threadIdx.x;
  if (i >= Npts * 32) return;
  int p = i >> 5, c = i & 31;
  out[i] = f2bf((c < 7) ? poly[(size_t)p * 7 + c] : 0.f);
}

// ---------------------------------------------------------------------------
// top-k (k=32) nearest kv per query; stable insertion matches lax.top_k ties
// ---------------------------------------------------------------------------
__global__ void topk_kernel(const float* __restrict__ qp,
                            const float* __restrict__ kvp,
                            const unsigned char* __restrict__ mask,
                            int Nq, int Nkv, int* __restrict__ out)
{
  int i = blockIdx.x * blockDim.x + threadIdx.x;
  if (i >= BB * Nq) return;
  int b = i / Nq;
  float qx = qp[(size_t)i * 2], qy = qp[(size_t)i * 2 + 1];
  float best[SPK]; int bidx[SPK];
#pragma unroll
  for (int j = 0; j < SPK; ++j) { best[j] = 3.4e38f; bidx[j] = 0; }
  for (int c = 0; c < Nkv; ++c) {
    float dx = qx - kvp[((size_t)b * Nkv + c) * 2];
    float dy = qy - kvp[((size_t)b * Nkv + c) * 2 + 1];
    float d2 = dx * dx + dy * dy;
    if (!mask[(size_t)b * Nkv + c]) d2 = 3.4e38f;
    if (d2 < best[SPK - 1]) {
      int pos = SPK - 1;
      while (pos > 0 && best[pos - 1] > d2) {
        best[pos] = best[pos - 1]; bidx[pos] = bidx[pos - 1]; --pos;
      }
      best[pos] = d2; bidx[pos] = c;
    }
  }
  for (int j = 0; j < SPK; ++j) out[(size_t)i * SPK + j] = bidx[j];
}

// ---------------------------------------------------------------------------
// Sparse attention with relative-position rotary phase.
// One wave32 per (b, query, head). Output written as bf16 (feeds o-proj GEMM).
// ---------------------------------------------------------------------------
__global__ void __launch_bounds__(256) sparse_attn_kernel(
    const float* __restrict__ Q, const float* __restrict__ Kb,
    const float* __restrict__ Vb, const int* __restrict__ idx,
    const float* __restrict__ qp, const float* __restrict__ kvp,
    const float* __restrict__ qh, const float* __restrict__ kvh,
    const unsigned char* __restrict__ mask,
    int Nq, int Nkv, u16* __restrict__ out)
{
  __shared__ float attn_s[8][SPK];
  __shared__ int   row_s[8][SPK];
  const int lane = threadIdx.x & 31;
  const int wv   = threadIdx.x >> 5;
  const int gid  = blockIdx.x * 8 + wv;
  const int total = BB * Nq * NHEADS;
  const bool active = gid < total;

  int h = 0, qrow = 0, kvrow = 0;
  float score = -1e9f;
  if (active) {
    int b = gid / (Nq * NHEADS);
    int r = gid % (Nq * NHEADS);
    int n = r / NHEADS; h = r % NHEADS;
    qrow = b * Nq + n;
    int ii = idx[(size_t)qrow * SPK + lane];
    kvrow = b * Nkv + ii;
    float rx = kvp[(size_t)kvrow * 2]     - qp[(size_t)qrow * 2];
    float ry = kvp[(size_t)kvrow * 2 + 1] - qp[(size_t)qrow * 2 + 1];
    float rh = kvh[kvrow] - qh[qrow];
    const float* qv = Q  + (size_t)qrow  * DMODEL + h * DHEAD;
    const float* kv = Kb + (size_t)kvrow * DMODEL + h * DHEAD;
    float s = 0.f;
#pragma unroll
    for (int d = 0; d < 16; ++d) {
      float invf = __powf(10000.f, -(float)d / 16.f);
      float ang = ((d < 8) ? rx : ry) * invf + rh;
      float sn, cs;
      __sincosf(ang, &sn, &cs);
      float k1 = kv[d], k2 = kv[d + 16];
      s += qv[d]      * (k1 * cs - k2 * sn)
         + qv[d + 16] * (k1 * sn + k2 * cs);
    }
    s *= 0.17677669529663687f;   // 1/sqrt(Dh)
    if (!mask[kvrow]) s = -1e9f;
    score = s;
  }
  float mx = score;
  for (int m = 16; m > 0; m >>= 1) mx = fmaxf(mx, __shfl_xor(mx, m, 32));
  float e = active ? __expf(score - mx) : 0.f;
  float sm = e;
  for (int m = 16; m > 0; m >>= 1) sm += __shfl_xor(sm, m, 32);
  attn_s[wv][lane] = e / sm;
  row_s[wv][lane]  = kvrow;
  __syncthreads();
  if (active) {
    float acc = 0.f;
#pragma unroll
    for (int j = 0; j < SPK; ++j)
      acc += attn_s[wv][j] * Vb[(size_t)row_s[wv][j] * DMODEL + h * DHEAD + lane];
    out[(size_t)qrow * DMODEL + h * DHEAD + lane] = f2bf(acc);
  }
}

// ---------------------------------------------------------------------------
// out = LayerNorm(xa + xb) * g + beta   (wave32 per token, D=256)
// Writes f32 and optionally bf16.
// ---------------------------------------------------------------------------
__global__ void __launch_bounds__(256) resid_ln_kernel(
    const float* __restrict__ xa, const float* __restrict__ xb,
    const float* __restrict__ g, const float* __restrict__ beta,
    float* __restrict__ outF, u16* __restrict__ outH, int Ntok)
{
  const int lane = threadIdx.x & 31;
  const int wv   = threadIdx.x >> 5;
  const int tok  = blockIdx.x * 8 + wv;
  if (tok >= Ntok) return;
  float v[8];
  float s = 0.f;
#pragma unroll
  for (int e = 0; e < 8; ++e) {
    int c = lane + 32 * e;
    v[e] = xa[(size_t)tok * DMODEL + c] + xb[(size_t)tok * DMODEL + c];
    s += v[e];
  }
  for (int m = 16; m > 0; m >>= 1) s += __shfl_xor(s, m, 32);
  float mean = s * (1.f / 256.f);
  float q = 0.f;
#pragma unroll
  for (int e = 0; e < 8; ++e) { float d = v[e] - mean; q += d * d; }
  for (int m = 16; m > 0; m >>= 1) q += __shfl_xor(q, m, 32);
  float rstd = rsqrtf(q * (1.f / 256.f) + 1e-5f);
#pragma unroll
  for (int e = 0; e < 8; ++e) {
    int c = lane + 32 * e;
    float o = (v[e] - mean) * rstd * g[c] + beta[c];
    outF[(size_t)tok * DMODEL + c] = o;
    if (outH) outH[(size_t)tok * DMODEL + c] = f2bf(o);
  }
}

__global__ void copy_out_kernel(const float* __restrict__ ag,
                                const float* __restrict__ mp,
                                float* __restrict__ out)
{
  int i = blockIdx.x * blockDim.x + threadIdx.x;
  const int nA = AG_TOK * DMODEL;
  const int nM = MP_TOK * DMODEL;
  if (i < nA) out[i] = ag[i];
  else if (i < nA + nM) out[i] = mp[i - nA];
}

// ---------------------------------------------------------------------------
// Host orchestration
// ---------------------------------------------------------------------------
extern "C" void kernel_launch(void* const* d_in, const int* in_sizes, int n_in,
                              void* d_out, int out_size, void* d_ws, size_t ws_size,
                              hipStream_t stream)
{
  (void)in_sizes; (void)n_in; (void)out_size; (void)ws_size;

  const float* obj_trajs      = (const float*)d_in[0];
  const float* obj_trajs_mask = (const float*)d_in[1];
  const float* obj_positions  = (const float*)d_in[2];
  const float* obj_headings   = (const float*)d_in[3];
  const unsigned char* agent_mask = (const unsigned char*)d_in[4];
  const float* map_polylines  = (const float*)d_in[5];
  const float* map_poly_mask  = (const float*)d_in[6];
  const float* map_center     = (const float*)d_in[7];
  const float* map_headings   = (const float*)d_in[8];
  const unsigned char* map_mask = (const unsigned char*)d_in[9];

  // params flattened jax.tree_util-style (sorted dict keys, lists in order):
  //  10..17  agent_enc: out.W,out.b, post0.W,b, post1.W,b, pre0.W,b
  //  18+48*i layer i: aa(16), am(16), mm(16); block leaves:
  //    k.W,k.b,o.W,o.b,q.W,q.b,v.W,v.b,l1.W,l1.b,l2.W,l2.b,n1.b,n1.g,n2.b,n2.g
  //  306..317 map_enc: out.W,b, post0.W,b, post1.W,b, pre0.W,b, pre1.W,b, pre2.W,b
  auto prm = [&](int i) { return (const float*)d_in[i]; };

  // ---- workspace bump allocator ----
  char* base = (char*)d_ws;
  size_t cur = 0;
  auto alloc = [&](size_t bytes) {
    cur = (cur + 255) & ~(size_t)255;
    void* p = base + cur;
    cur += bytes;
    return p;
  };

  // persistent buffers
  float* map_feat_f   = (float*)alloc((size_t)MP_TOK * DMODEL * 4);
  float* agent_feat_f = (float*)alloc((size_t)AG_TOK * DMODEL * 4);
  u16*   map_feat_h   = (u16*)  alloc((size_t)MP_TOK * DMODEL * 2);
  u16*   agent_feat_h = (u16*)  alloc((size_t)AG_TOK * DMODEL * 2);
  int*   idx_mm = (int*)alloc((size_t)BB * MM_N * SPK * 4);
  int*   idx_aa = (int*)alloc((size_t)BB * AA_N * SPK * 4);
  int*   idx_am = (int*)alloc((size_t)BB * AA_N * SPK * 4);

  // ---- weight packing (once per launch) ----
  u16* pw[320];
  auto packW = [&](int idx, int K, int N) {
    int KB = (K + 31) >> 5;
    size_t elems = (size_t)(N / 16) * KB * 512;
    u16* p = (u16*)alloc(elems * 2);
    pack_w_kernel<<<((int)elems + 255) / 256, 256, 0, stream>>>(
        prm(idx), K, N, p, (int)elems);
    pw[idx] = p;
  };
  packW(16, 11, 256);  packW(12, 512, 256);  packW(14, 256, 256);  packW(10, 256, 256);
  packW(312, 7, 64);   packW(314, 64, 64);   packW(316, 64, 64);
  packW(308, 128, 64); packW(310, 64, 64);   packW(306, 64, 256);
  for (int i = 0; i < 6; ++i) {
    int L = 18 + 48 * i;
    for (int blk = 0; blk < 3; ++blk) {
      int pb = L + 16 * blk;
      packW(pb + 0, 256, 256);  packW(pb + 2, 256, 256);
      packW(pb + 4, 256, 256);  packW(pb + 6, 256, 256);
      packW(pb + 8, 256, 1024); packW(pb + 10, 1024, 256);
    }
  }

  // phase-overlaid scratch starts here
  size_t Sbase = (cur + 255) & ~(size_t)255;

  auto gemm = [&](const u16* A1, int K1, const u16* A2, int K2, int rowDiv,
                  const u16* Wp, const float* bias, float* outF, u16* outH,
                  int M, int N, int relu) {
    dim3 grid(M / 16, N / 64);
    gemm_kernel<<<grid, 32, 0, stream>>>(A1, K1, A2, K2, rowDiv, Wp, bias,
                                         outF, outH, N, relu);
  };

  // ---------------- Agent PointNet ----------------
  {
    cur = Sbase;
    u16* agIn  = (u16*)alloc((size_t)AG_PTS * 32 * 2);
    u16* A0    = (u16*)alloc((size_t)AG_PTS * DMODEL * 2);
    u16* A1b   = (u16*)alloc((size_t)AG_PTS * DMODEL * 2);
    u16* pool  = (u16*)alloc((size_t)AG_TOK * DMODEL * 2);
    agent_input_kernel<<<(AG_PTS * 32 + 255) / 256, 256, 0, stream>>>(
        obj_trajs, obj_trajs_mask, agIn, AG_PTS);
    gemm(agIn, 32, nullptr, 0, 1, pw[16], prm(17), nullptr, A0, AG_PTS, 256, 1);
    masked_max_kernel<<<(AG_TOK * 256 + 255) / 256, 256, 0, stream>>>(
        A0, obj_trajs_mask, pool, AG_TOK, TT, 256);
    gemm(A0, 256, pool, 256, TT, pw[12], prm(13), nullptr, A1b, AG_PTS, 256, 1);
    gemm(A1b, 256, nullptr, 0, 1, pw[14], prm(15), nullptr, A0, AG_PTS, 256, 1);
    masked_max_kernel<<<(AG_TOK * 256 + 255) / 256, 256, 0, stream>>>(
        A0, obj_trajs_mask, pool, AG_TOK, TT, 256);
    gemm(pool, 256, nullptr, 0, 1, pw[10], prm(11),
         agent_feat_f, agent_feat_h, AG_TOK, 256, 0);
  }

  // ---------------- Map PointNet ----------------
  {
    cur = Sbase;
    u16* mpIn = (u16*)alloc((size_t)MP_PTS * 32 * 2);
    u16* B0   = (u16*)alloc((size_t)MP_PTS * 64 * 2);
    u16* B1   = (u16*)alloc((size_t)MP_PTS * 64 * 2);
    u16* pool = (u16*)alloc((size_t)MP_TOK * 64 * 2);
    map_input_kernel<<<(MP_PTS * 32 + 255) / 256, 256, 0, stream>>>(
        map_polylines, mpIn, MP_PTS);
    gemm(mpIn, 32, nullptr, 0, 1, pw[312], prm(313), nullptr, B0, MP_PTS, 64, 1);
    gemm(B0, 64, nullptr, 0, 1, pw[314], prm(315), nullptr, B1, MP_PTS, 64, 1);
    gemm(B1, 64, nullptr, 0, 1, pw[316], prm(317), nullptr, B0, MP_PTS, 64, 1);
    masked_max_kernel<<<(MP_TOK * 64 + 255) / 256, 256, 0, stream>>>(
        B0, map_poly_mask, pool, MP_TOK, PP, 64);
    gemm(B0, 64, pool, 64, PP, pw[308], prm(309), nullptr, B1, MP_PTS, 64, 1);
    gemm(B1, 64, nullptr, 0, 1, pw[310], prm(311), nullptr, B0, MP_PTS, 64, 1);
    masked_max_kernel<<<(MP_TOK * 64 + 255) / 256, 256, 0, stream>>>(
        B0, map_poly_mask, pool, MP_TOK, PP, 64);
    gemm(pool, 64, nullptr, 0, 1, pw[306], prm(307),
         map_feat_f, map_feat_h, MP_TOK, 256, 0);
  }

  // ---------------- top-k neighbor selection ----------------
  topk_kernel<<<(BB * MM_N + 255) / 256, 256, 0, stream>>>(
      map_center, map_center, map_mask, MM_N, MM_N, idx_mm);
  topk_kernel<<<(BB * AA_N + 255) / 256, 256, 0, stream>>>(
      obj_positions, obj_positions, agent_mask, AA_N, AA_N, idx_aa);
  topk_kernel<<<(BB * AA_N + 255) / 256, 256, 0, stream>>>(
      obj_positions, map_center, map_mask, AA_N, MM_N, idx_am);

  // ---------------- transformer blocks ----------------
  cur = Sbase;
  float* qbuf = (float*)alloc((size_t)MP_TOK * DMODEL * 4);
  float* kbuf = (float*)alloc((size_t)MP_TOK * DMODEL * 4);
  float* vbuf = (float*)alloc((size_t)MP_TOK * DMODEL * 4);
  u16*   abuf = (u16*)  alloc((size_t)MP_TOK * DMODEL * 2);
  float* x1f  = (float*)alloc((size_t)MP_TOK * DMODEL * 4);
  u16*   x1h  = (u16*)  alloc((size_t)MP_TOK * DMODEL * 2);
  float* fob  = (float*)alloc((size_t)MP_TOK * DMODEL * 4);
  u16*   fhb  = (u16*)  alloc((size_t)MP_TOK * 1024 * 2);

  auto run_block = [&](float* qfF, u16* qfH, u16* kvH, int NqTok, int NkvTok,
                       int NqB, int NkvB,
                       const float* qpv, const float* kvpv,
                       const float* qhv, const float* kvhv,
                       const unsigned char* msk, const int* idx, int pb) {
    gemm(qfH, 256, nullptr, 0, 1, pw[pb + 4], prm(pb + 5), qbuf, nullptr, NqTok, 256, 0);
    gemm(kvH, 256, nullptr, 0, 1, pw[pb + 0], prm(pb + 1), kbuf, nullptr, NkvTok, 256, 0);
    gemm(kvH, 256, nullptr, 0, 1, pw[pb + 6], prm(pb + 7), vbuf, nullptr, NkvTok, 256, 0);
    int totalw = BB * NqB * NHEADS;
    sparse_attn_kernel<<<(totalw + 7) / 8, 256, 0, stream>>>(
        qbuf, kbuf, vbuf, idx, qpv, kvpv, qhv, kvhv, msk, NqB, NkvB, abuf);
    gemm(abuf, 256, nullptr, 0, 1, pw[pb + 2], prm(pb + 3), fob, nullptr, NqTok, 256, 0);
    resid_ln_kernel<<<(NqTok + 7) / 8, 256, 0, stream>>>(
        qfF, fob, prm(pb + 13), prm(pb + 12), x1f, x1h, NqTok);
    gemm(x1h, 256, nullptr, 0, 1, pw[pb + 8], prm(pb + 9), nullptr, fhb, NqTok, 1024, 1);
    gemm(fhb, 1024, nullptr, 0, 1, pw[pb + 10], prm(pb + 11), fob, nullptr, NqTok, 256, 0);
    resid_ln_kernel<<<(NqTok + 7) / 8, 256, 0, stream>>>(
        x1f, fob, prm(pb + 15), prm(pb + 14), qfF, qfH, NqTok);
  };

  for (int i = 0; i < 6; ++i) {
    int L = 18 + 48 * i;
    // mm: map self-attention
    run_block(map_feat_f, map_feat_h, map_feat_h, MP_TOK, MP_TOK, MM_N, MM_N,
              map_center, map_center, map_headings, map_headings,
              map_mask, idx_mm, L + 32);
    // aa: agent self-attention
    run_block(agent_feat_f, agent_feat_h, agent_feat_h, AG_TOK, AG_TOK, AA_N, AA_N,
              obj_positions, obj_positions, obj_headings, obj_headings,
              agent_mask, idx_aa, L + 0);
    // am: agent -> map cross-attention
    run_block(agent_feat_f, agent_feat_h, map_feat_h, AG_TOK, MP_TOK, AA_N, MM_N,
              obj_positions, map_center, obj_headings, map_headings,
              map_mask, idx_am, L + 16);
  }

  const int nOut = AG_TOK * DMODEL + MP_TOK * DMODEL;
  copy_out_kernel<<<(nOut + 255) / 256, 256, 0, stream>>>(
      agent_feat_f, map_feat_f, (float*)d_out);
}
